// TransformerBlock_56478819942862
// MI455X (gfx1250) — compile-verified, matrix operand order checked by reading, not run on hardware
//
#include <hip/hip_runtime.h>
#include <hip/hip_bf16.h>
#include <cstdint>

#define VOCAB  32000
#define DMODEL 1024
#define NH     16
#define DHEAD  64
#define DFF    4096
#define BATCH  4
#define SEQ    2048
#define MTOT   (BATCH * SEQ)   // 8192

typedef __bf16 bf16;
typedef __attribute__((ext_vector_type(16))) __bf16 v16bf;
typedef __attribute__((ext_vector_type(8)))  float  v8f;
typedef int v4i __attribute__((vector_size(16)));   // matches builtin's V4i

#define AS1 __attribute__((address_space(1)))
#define AS3 __attribute__((address_space(3)))

union FragBF {
    v16bf  v;
    float4 q[2];
    __bf16 e[16];
};

__device__ inline v8f zero_v8f() {
    v8f z;
#pragma unroll
    for (int i = 0; i < 8; ++i) z[i] = 0.0f;
    return z;
}

__device__ inline v8f wmma_bf16(v16bf a, v16bf b, v8f c) {
    return __builtin_amdgcn_wmma_f32_16x16x32_bf16(
        /*neg_a=*/false, a, /*neg_b=*/false, b,
        /*c_mod=*/(short)0, c, /*reuse_a=*/false, /*reuse_b=*/false);
}

// ---------------------------------------------------------------------------
// CDNA5 async global->LDS copy (16B per lane) + ASYNCcnt wait
// ---------------------------------------------------------------------------
__device__ inline void async_copy_b128(void* lds_ptr, const void* gptr) {
#if __has_builtin(__builtin_amdgcn_global_load_async_to_lds_b128)
    __builtin_amdgcn_global_load_async_to_lds_b128(
        (AS1 v4i*)(void*)gptr, (AS3 v4i*)lds_ptr, 0, 0);
#else
    unsigned loff = (unsigned)(uintptr_t)lds_ptr;   // low 32b of flat LDS addr = LDS offset
    asm volatile("global_load_async_to_lds_b128 %0, %1, off"
                 :: "v"(loff), "v"(gptr) : "memory");
#endif
}

template <int N>
__device__ inline void wait_async() {
#if __has_builtin(__builtin_amdgcn_s_wait_asynccnt)
    __builtin_amdgcn_s_wait_asynccnt(N);
#else
    asm volatile("s_wait_asynccnt %0" :: "i"(N) : "memory");
#endif
}

// ---------------------------------------------------------------------------
// NT fragment load straight from global (used by attention).
//   lanes 0-15 : row = lane,     K = kt+[0..7]  and kt+[16..23]
//   lanes 16-31: row = lane-16,  K = kt+[8..15] and kt+[24..31]
// ---------------------------------------------------------------------------
__device__ inline v16bf load_frag_nt(const bf16* __restrict__ base, int ld, int kt) {
    const int lane = threadIdx.x & 31;
    const bf16* p = base + (size_t)(lane & 15) * (size_t)ld + kt + ((lane >> 4) << 3);
    FragBF u;
    u.q[0] = *(const float4*)(p);
    u.q[1] = *(const float4*)(p + 16);
    return u.v;
}

// Same NT fragment pattern, but from an LDS tile with 80-byte row pitch
// (64B of K data + 16B pad to spread DS banks).
__device__ inline v16bf lds_frag(const char* base, int row0) {
    const int lane = threadIdx.x & 31;
    const char* p = base + (size_t)(row0 + (lane & 15)) * 80 + ((lane >> 4) << 4);
    FragBF u;
    u.q[0] = *(const float4*)(p);        // K elems 0..7 (lo) / 8..15 (hi)
    u.q[1] = *(const float4*)(p + 32);   // K elems 16..23 (lo) / 24..31 (hi)
    return u.v;
}

// ---------------------------------------------------------------------------
// fp32 -> bf16 conversion (weights)
// ---------------------------------------------------------------------------
__global__ void k_f32_to_bf16(const float* __restrict__ src, bf16* __restrict__ dst, int n) {
    int i = blockIdx.x * 256 + threadIdx.x;
    if (i < n) dst[i] = (bf16)src[i];
}

// ---------------------------------------------------------------------------
// Embedding gather + sinusoidal positional encoding (faithful to reference)
// ---------------------------------------------------------------------------
__global__ void k_embed(const int* __restrict__ tok, const float* __restrict__ emb,
                        float* __restrict__ xf, bf16* __restrict__ xb) {
    size_t i = (size_t)blockIdx.x * 256 + threadIdx.x;   // [0, MTOT*DMODEL)
    int d = (int)(i % DMODEL);
    size_t row = i / DMODEL;                             // b*SEQ + s
    int s = (int)(row % SEQ);
    int de = d & ~1;                                     // even index of pair
    float angle = (float)s * __powf(10000.0f, -2.0f * (float)de / (float)DMODEL);
    float pe = (d & 1) ? __cosf(angle) : __sinf(angle);
    float x = emb[(size_t)tok[row] * DMODEL + d] + pe;
    xf[i] = x;
    xb[i] = (bf16)x;
}

// ---------------------------------------------------------------------------
// Generic NT WMMA GEMM:  C[M,N] = act(A[M,K] @ W[N,K]^T + bias[N])
// Block = 8 waves = 64M x 256N; wave tile 32M x 64N (8 accumulators).
// K staged in 32-wide slabs through double-buffered LDS using the CDNA5
// async global->LDS path (ASYNCcnt), fragments read back via ds_load_b128.
// Per K-step, per wave: 5 async copies, wait(<=5), barrier, 12 ds loads,
// 8 WMMAs, barrier.  All dims divide evenly -> no bounds checks, EXEC all 1s.
// ---------------------------------------------------------------------------
#define TILE_A_BYTES (64 * 80)     // 64 rows x (64B data + 16B pad)
#define TILE_W_BYTES (256 * 80)    // 256 rows
#define TILE_BYTES   (TILE_A_BYTES + TILE_W_BYTES)

__global__ __launch_bounds__(256)
void k_gemm_nt(const bf16* __restrict__ A, const bf16* __restrict__ W,
               const float* __restrict__ bias,
               float* __restrict__ Cf, bf16* __restrict__ Cb,
               int M, int N, int K, int relu) {
    __shared__ __align__(16) char smem[2 * TILE_BYTES];   // 51200 B

    const int tid  = threadIdx.x;
    const int lane = tid & 31;
    const int wid  = tid >> 5;
    const int mblk = blockIdx.y * 64;
    const int nblk = blockIdx.x * 256;
    const int m0 = (wid & 1) * 32;          // wave tile origin inside block tile
    const int n0 = (wid >> 1) * 64;

    const int crow   = tid >> 2;            // copy row 0..63
    const int cchunk = (tid & 3) * 16;      // 16B chunk within 64B row

    auto issue = [&](int buf, int kt) {
        char* base = smem + buf * TILE_BYTES;
        // A tile: 64 rows x 32 K
        async_copy_b128(base + (size_t)crow * 80 + cchunk,
                        A + (size_t)(mblk + crow) * K + kt + (tid & 3) * 8);
        // W tile: 256 rows x 32 K
        char* wbse = base + TILE_A_BYTES;
#pragma unroll
        for (int j = 0; j < 4; ++j) {
            int row = crow + j * 64;
            async_copy_b128(wbse + (size_t)row * 80 + cchunk,
                            W + (size_t)(nblk + row) * K + kt + (tid & 3) * 8);
        }
    };

    v8f acc[2][4];
#pragma unroll
    for (int i = 0; i < 2; ++i)
#pragma unroll
        for (int j = 0; j < 4; ++j) acc[i][j] = zero_v8f();

    issue(0, 0);
    int buf = 0;
    for (int kt = 0; kt < K; kt += 32) {
        bool more = (kt + 32) < K;
        if (more) {
            issue(buf ^ 1, kt + 32);   // prefetch next slab into other buffer
            wait_async<5>();           // previous group (current buffer) done
        } else {
            wait_async<0>();
        }
        __syncthreads();               // all waves' copies for this buffer visible

        const char* base = smem + buf * TILE_BYTES;
        const char* wbse = base + TILE_A_BYTES;
        v16bf a0 = lds_frag(base, m0);
        v16bf a1 = lds_frag(base, m0 + 16);
#pragma unroll
        for (int j = 0; j < 4; ++j) {
            v16bf w = lds_frag(wbse, n0 + j * 16);
            acc[0][j] = wmma_bf16(a0, w, acc[0][j]);
            acc[1][j] = wmma_bf16(a1, w, acc[1][j]);
        }
        __syncthreads();               // done reading: buffer may be overwritten
        buf ^= 1;
    }

    const int ncol = lane & 15;
    const int rhi  = (lane >> 4) << 3;
#pragma unroll
    for (int j = 0; j < 4; ++j) {
        int n = nblk + n0 + j * 16 + ncol;
        float bv = bias ? bias[n] : 0.0f;
#pragma unroll
        for (int i = 0; i < 2; ++i)
#pragma unroll
            for (int r = 0; r < 8; ++r) {
                int m = mblk + m0 + i * 16 + r + rhi;
                float y = acc[i][j][r] + bv;
                if (relu) y = fmaxf(y, 0.0f);
                size_t off = (size_t)m * N + n;
                if (Cf) Cf[off] = y;
                if (Cb) Cb[off] = (bf16)y;
            }
    }
}

// ---------------------------------------------------------------------------
// QKV split: qkv[B,S,3D] -> Q[B,H,S,DH] (pre-scaled by 1/sqrt(DH)),
//            K[B,H,S,DH], Vt[B,H,DH,S] (transposed for the PV matmul)
// ---------------------------------------------------------------------------
__global__ void k_split(const bf16* __restrict__ qkv, bf16* __restrict__ Q,
                        bf16* __restrict__ Km, bf16* __restrict__ Vt) {
    size_t i = (size_t)blockIdx.x * 256 + threadIdx.x;   // [0, MTOT*DMODEL)
    int d = (int)(i % DHEAD);
    size_t t = i / DHEAD;
    int h = (int)(t % NH);
    size_t r = t / NH;                                    // b*SEQ + s
    int s = (int)(r % SEQ);
    int b = (int)(r / SEQ);
    size_t qoff = r * (size_t)(3 * DMODEL) + h * DHEAD + d;
    size_t bh = (size_t)(b * NH + h);
    size_t o1 = (bh * SEQ + s) * DHEAD + d;
    Q[o1]  = (bf16)((float)qkv[qoff] * 0.125f);           // 1/sqrt(64)
    Km[o1] = qkv[qoff + DMODEL];
    Vt[(bh * DHEAD + d) * SEQ + s] = qkv[qoff + 2 * DMODEL];
}

// ---------------------------------------------------------------------------
// Flash attention, one wave per (b, h, 16-row q-tile).
// Scores computed TRANSPOSED (S^T = K @ Q^T): each lane owns one query column
// (q = lane&15) -> softmax stats are per-lane scalars + one shfl_xor(16), and
// exp(S^T) lands directly in the B-fragment layout for ctx^T = V^T @ P^T.
// ---------------------------------------------------------------------------
__global__ __launch_bounds__(256)
void k_attn(const bf16* __restrict__ Qm, const bf16* __restrict__ Km,
            const bf16* __restrict__ Vt, bf16* __restrict__ ctx) {
    const int lane = threadIdx.x & 31;
    const int wid  = threadIdx.x >> 5;
    const int gw   = blockIdx.x * 8 + wid;     // [0, B*H*S/16)
    const int qtiles = SEQ / 16;               // 128
    const int qt = gw % qtiles;
    const int bh = gw / qtiles;                // [0, 64)
    const int b = bh / NH, h = bh % NH;

    const bf16* qbase = Qm + ((size_t)bh * SEQ + qt * 16) * DHEAD;
    v16bf q0 = load_frag_nt(qbase, DHEAD, 0);    // d = 0..31
    v16bf q1 = load_frag_nt(qbase, DHEAD, 32);   // d = 32..63

    v8f acc[4];                                   // ctx^T, 4 d-tiles of 16
#pragma unroll
    for (int t = 0; t < 4; ++t) acc[t] = zero_v8f();
    float m_i = -3.0e38f, l_i = 0.0f;

    for (int kk = 0; kk < SEQ; kk += 32) {
        const bf16* kb0 = Km + ((size_t)bh * SEQ + kk) * DHEAD;
        const bf16* kb1 = kb0 + 16 * DHEAD;
        v8f s0 = zero_v8f();                      // S^T tile, keys kk..kk+15
        s0 = wmma_bf16(load_frag_nt(kb0, DHEAD, 0),  q0, s0);
        s0 = wmma_bf16(load_frag_nt(kb0, DHEAD, 32), q1, s0);
        v8f s1 = zero_v8f();                      // keys kk+16..kk+31
        s1 = wmma_bf16(load_frag_nt(kb1, DHEAD, 0),  q0, s1);
        s1 = wmma_bf16(load_frag_nt(kb1, DHEAD, 32), q1, s1);

        float mloc = -3.0e38f;
#pragma unroll
        for (int r = 0; r < 8; ++r) {
            mloc = fmaxf(mloc, s0[r]);
            mloc = fmaxf(mloc, s1[r]);
        }
        mloc = fmaxf(mloc, __shfl_xor(mloc, 16, 32));
        float m_new = fmaxf(m_i, mloc);
        float alpha = __expf(m_i - m_new);

        FragBF p;                                 // P^T as B fragment (K=32)
        float lloc = 0.0f;
#pragma unroll
        for (int r = 0; r < 8; ++r) {
            float e0 = __expf(s0[r] - m_new);
            float e1 = __expf(s1[r] - m_new);
            lloc += e0 + e1;
            p.e[r]     = (bf16)e0;                // k_local 0..7  (lo) / 8..15 (hi)
            p.e[r + 8] = (bf16)e1;                // k_local 16..23 (lo)/ 24..31 (hi)
        }
        lloc += __shfl_xor(lloc, 16, 32);
        l_i = l_i * alpha + lloc;
        m_i = m_new;

#pragma unroll
        for (int t = 0; t < 4; ++t) {
#pragma unroll
            for (int r = 0; r < 8; ++r) acc[t][r] *= alpha;   // per-q rescale
            const bf16* vb = Vt + ((size_t)bh * DHEAD + t * 16) * SEQ;
            acc[t] = wmma_bf16(load_frag_nt(vb, SEQ, kk), p.v, acc[t]);
        }
    }

    float invl = 1.0f / l_i;
    const int qcol = lane & 15;
    const int rhi  = (lane >> 4) << 3;
#pragma unroll
    for (int t = 0; t < 4; ++t)
#pragma unroll
        for (int r = 0; r < 8; ++r) {
            int dloc = t * 16 + r + rhi;
            size_t off = ((size_t)b * SEQ + qt * 16 + qcol) * DMODEL + h * DHEAD + dloc;
            ctx[off] = (bf16)(acc[t][r] * invl);
        }
}

// ---------------------------------------------------------------------------
// Fused residual add + LayerNorm. One block (256 thr) per 1024-wide row.
// ---------------------------------------------------------------------------
__global__ __launch_bounds__(256)
void k_add_ln(const float* __restrict__ A, const float* __restrict__ Bm,
              const float* __restrict__ g, const float* __restrict__ beta,
              float* __restrict__ outf, bf16* __restrict__ outb) {
    __shared__ float ss[256], sq[256];
    const int row = blockIdx.x;
    const int tid = threadIdx.x;
    const float* pa = A  + (size_t)row * DMODEL;
    const float* pb = Bm + (size_t)row * DMODEL;
    float v[4];
    float s = 0.0f, s2 = 0.0f;
#pragma unroll
    for (int i = 0; i < 4; ++i) {
        int c = tid + i * 256;
        float x = pa[c] + pb[c];
        v[i] = x; s += x; s2 += x * x;
    }
    ss[tid] = s; sq[tid] = s2;
    __syncthreads();
    for (int o = 128; o > 0; o >>= 1) {
        if (tid < o) { ss[tid] += ss[tid + o]; sq[tid] += sq[tid + o]; }
        __syncthreads();
    }
    float mu  = ss[0] * (1.0f / DMODEL);
    float var = sq[0] * (1.0f / DMODEL) - mu * mu;
    float rs  = rsqrtf(var + 1e-5f);
#pragma unroll
    for (int i = 0; i < 4; ++i) {
        int c = tid + i * 256;
        float y = (v[i] - mu) * rs * g[c] + beta[c];
        size_t off = (size_t)row * DMODEL + c;
        outf[off] = y;
        if (outb) outb[off] = (bf16)y;
    }
}

// ---------------------------------------------------------------------------
// Host driver
// ---------------------------------------------------------------------------
extern "C" void kernel_launch(void* const* d_in, const int* in_sizes, int n_in,
                              void* d_out, int out_size, void* d_ws, size_t ws_size,
                              hipStream_t stream) {
    const int*   tokens     = (const int*)  d_in[0];
    const float* emb        = (const float*)d_in[1];
    const float* in_proj_w  = (const float*)d_in[2];
    const float* in_proj_b  = (const float*)d_in[3];
    const float* out_proj_w = (const float*)d_in[4];
    const float* out_proj_b = (const float*)d_in[5];
    const float* w1         = (const float*)d_in[6];
    const float* b1         = (const float*)d_in[7];
    const float* w2         = (const float*)d_in[8];
    const float* b2         = (const float*)d_in[9];
    const float* g1         = (const float*)d_in[10];
    const float* beta1      = (const float*)d_in[11];
    const float* g2         = (const float*)d_in[12];
    const float* beta2      = (const float*)d_in[13];
    float* out = (float*)d_out;

    char* ws = (char*)d_ws;
    size_t off = 0;
    auto alloc = [&](size_t bytes) -> void* {
        void* p = ws + off;
        off = (off + bytes + 255) & ~(size_t)255;
        return p;
    };

    float* xf   = (float*)alloc((size_t)MTOT * DMODEL * 4);
    bf16*  xb   = (bf16*) alloc((size_t)MTOT * DMODEL * 2);
    bf16*  wq   = (bf16*) alloc((size_t)3 * DMODEL * DMODEL * 2);
    bf16*  wo   = (bf16*) alloc((size_t)DMODEL * DMODEL * 2);
    bf16*  ww1  = (bf16*) alloc((size_t)DFF * DMODEL * 2);
    bf16*  ww2  = (bf16*) alloc((size_t)DMODEL * DFF * 2);
    bf16*  qkv  = (bf16*) alloc((size_t)MTOT * 3 * DMODEL * 2);
    bf16*  Q    = (bf16*) alloc((size_t)MTOT * DMODEL * 2);
    bf16*  Km   = (bf16*) alloc((size_t)MTOT * DMODEL * 2);
    bf16*  Vt   = (bf16*) alloc((size_t)MTOT * DMODEL * 2);
    bf16*  ctx  = (bf16*) alloc((size_t)MTOT * DMODEL * 2);
    float* attn = (float*)alloc((size_t)MTOT * DMODEL * 4);
    float* hf   = (float*)alloc((size_t)MTOT * DMODEL * 4);
    bf16*  hb   = (bf16*) alloc((size_t)MTOT * DMODEL * 2);
    bf16*  ff1  = (bf16*) alloc((size_t)MTOT * DFF * 2);
    float* ff2  = (float*)alloc((size_t)MTOT * DMODEL * 4);

    auto cgrid = [](int n) { return (n + 255) / 256; };
    k_f32_to_bf16<<<cgrid(3 * DMODEL * DMODEL), 256, 0, stream>>>(in_proj_w,  wq,  3 * DMODEL * DMODEL);
    k_f32_to_bf16<<<cgrid(DMODEL * DMODEL),     256, 0, stream>>>(out_proj_w, wo,  DMODEL * DMODEL);
    k_f32_to_bf16<<<cgrid(DFF * DMODEL),        256, 0, stream>>>(w1,         ww1, DFF * DMODEL);
    k_f32_to_bf16<<<cgrid(DMODEL * DFF),        256, 0, stream>>>(w2,         ww2, DMODEL * DFF);

    // x = emb[tokens] + pos_enc
    k_embed<<<(MTOT * DMODEL) / 256, 256, 0, stream>>>(tokens, emb, xf, xb);

    // qkv = x @ in_proj_w^T + in_proj_b
    k_gemm_nt<<<dim3((3 * DMODEL) / 256, MTOT / 64), 256, 0, stream>>>(
        xb, wq, in_proj_b, nullptr, qkv, MTOT, 3 * DMODEL, DMODEL, 0);

    // split into heads (Q pre-scaled, V transposed)
    k_split<<<(MTOT * DMODEL) / 256, 256, 0, stream>>>(qkv, Q, Km, Vt);

    // flash attention -> ctx[B,S,D] bf16
    k_attn<<<(BATCH * NH * (SEQ / 16)) / 8, 256, 0, stream>>>(Q, Km, Vt, ctx);

    // attn_out = ctx @ out_proj_w^T + out_proj_b
    k_gemm_nt<<<dim3(DMODEL / 256, MTOT / 64), 256, 0, stream>>>(
        ctx, wo, out_proj_b, attn, nullptr, MTOT, DMODEL, DMODEL, 0);

    // h = LN(x + attn_out)
    k_add_ln<<<MTOT, 256, 0, stream>>>(xf, attn, g1, beta1, hf, hb);

    // ff1 = relu(h @ w1^T + b1)
    k_gemm_nt<<<dim3(DFF / 256, MTOT / 64), 256, 0, stream>>>(
        hb, ww1, b1, nullptr, ff1, MTOT, DFF, DMODEL, 1);

    // ff2 = ff1 @ w2^T + b2
    k_gemm_nt<<<dim3(DMODEL / 256, MTOT / 64), 256, 0, stream>>>(
        ff1, ww2, b2, ff2, nullptr, MTOT, DMODEL, DFF, 0);

    // out = LN(h + ff2)
    k_add_ln<<<MTOT, 256, 0, stream>>>(hf, ff2, g2, beta2, out, nullptr);
}
